// BinaryDenseLayer_79044578116096
// MI455X (gfx1250) — compile-verified, matrix operand order checked by reading, not run on hardware
//
#include <hip/hip_runtime.h>

typedef __attribute__((ext_vector_type(16))) __bf16 v16bf;
typedef __attribute__((ext_vector_type(8)))  __bf16 v8bf;
typedef __attribute__((ext_vector_type(8)))  float  v8f;
typedef __attribute__((ext_vector_type(4)))  unsigned int u32x4;
typedef __attribute__((ext_vector_type(8)))  int i32x8;
typedef __attribute__((ext_vector_type(4)))  int i32x4;

#define BM 128
#define BN 256
#define KT 32
#define KT_PAD 40   // 80-byte LDS row stride; TDM pad: every 16 DWORDs insert 4 DWORDs

static __device__ __forceinline__ unsigned short f32_to_bf16_rne(float f) {
    unsigned int u = __builtin_bit_cast(unsigned int, f);
    unsigned int r = u + 0x7FFFu + ((u >> 16) & 1u);
    return (unsigned short)(r >> 16);
}

// LDS byte offset of a generic pointer to __shared__ data (AS(3) pointers are 32-bit)
static __device__ __forceinline__ unsigned lds_off_u32(const void* p) {
    return (unsigned)(unsigned long long)(const __attribute__((address_space(3))) char*)p;
}

// ---- Tensor Data Mover: 2-D tile load, memory -> LDS, with LDS row padding ----------
// D# per CDNA5 ISA 8.3/8.4: group0 = {count/type/lds_addr/global_addr},
// group1 = {data_size, pad ctl, tensor dims, tile dims, dim0 stride}; groups 2/3 zero.
static __device__ __forceinline__ void tdm_load_2d(
    unsigned lds_addr, unsigned long long gaddr,
    unsigned tensor_d0, unsigned tensor_d1,
    unsigned tile_d0, unsigned tile_d1,
    unsigned long long d0_stride)
{
    u32x4 g0;
    g0.x = 1u;                                              // count=1 (valid), user mode
    g0.y = lds_addr;                                        // lds_addr [63:32]
    g0.z = (unsigned)(gaddr & 0xFFFFFFFFu);                 // global_addr[31:0]
    g0.w = (unsigned)((gaddr >> 32) & 0x01FFFFFFu)          // global_addr[56:32]
         | (2u << 30);                                      // type=2 ("image")

    i32x8 g1;
    g1[0] = (int)((1u << 16)        // data_size=1 -> 2-byte elements
                | (1u << 20)        // pad_enable
                | (3u << 22)        // pad_interval code 3 = 16 DWORDs (one 64B row)
                | (3u << 25));      // pad_amount  code 3 = 4 DWORDs (16B) -> 80B stride
    g1[1] = (int)((tensor_d0 & 0xFFFFu) << 16);                         // dim0[15:0] @63:48
    g1[2] = (int)((tensor_d0 >> 16) | ((tensor_d1 & 0xFFFFu) << 16));   // dim0 hi / dim1 lo
    g1[3] = (int)((tensor_d1 >> 16) | ((tile_d0 & 0xFFFFu) << 16));     // dim1 hi / tile_dim0
    g1[4] = (int)(tile_d1 & 0xFFFFu);                                   // tile_dim1; tile_dim2=0
    g1[5] = (int)(unsigned)(d0_stride & 0xFFFFFFFFu);                   // dim0_stride[31:0]
    g1[6] = (int)(unsigned)((d0_stride >> 32) & 0xFFFFu);               // dim0_stride[47:32]
    g1[7] = 0;                                                          // dim1_stride unused (2D)

    i32x4 gz4 = {0, 0, 0, 0};                               // groups 2/3: 2-D tile only
    i32x8 gz8 = {0, 0, 0, 0, 0, 0, 0, 0};                   // extra group (6-arg form), unused
    __builtin_amdgcn_tensor_load_to_lds(g0, g1, gz4, gz4, gz8, 0);
}

// ---------------- Stage 1a: quantize W to bf16 {-1,0,+1}, transposed to N x K ----------
__global__ __launch_bounds__(256) void quantize_transpose_kernel(
    const float* __restrict__ W, unsigned short* __restrict__ Wq_t, int K, int N)
{
    __shared__ unsigned short tile[32][33];
    const int k0 = blockIdx.y * 32;
    const int n0 = blockIdx.x * 32;
    const int tx = threadIdx.x;       // 0..31
    const int ty = threadIdx.y;       // 0..7
    #pragma unroll
    for (int i = ty; i < 32; i += 8) {
        float w = W[(size_t)(k0 + i) * N + (n0 + tx)];
        unsigned short q = (w > 0.0f) ? 0x3F80u : ((w < 0.0f) ? 0xBF80u : 0u);
        tile[i][tx] = q;
    }
    __syncthreads();
    #pragma unroll
    for (int i = ty; i < 32; i += 8) {
        Wq_t[(size_t)(n0 + i) * K + (k0 + tx)] = tile[tx][i];
    }
}

// ---------------- Stage 1b: convert x (f32) -> bf16 ----------------------------------
__global__ __launch_bounds__(256) void convert_x_kernel(
    const float* __restrict__ x, unsigned short* __restrict__ xq, long n)
{
    long i = ((long)blockIdx.x * 256 + threadIdx.x) * 4;
    if (i + 3 < n) {
        float4 v = *(const float4*)(x + i);
        ushort4 q;
        q.x = f32_to_bf16_rne(v.x);
        q.y = f32_to_bf16_rne(v.y);
        q.z = f32_to_bf16_rne(v.z);
        q.w = f32_to_bf16_rne(v.w);
        *(ushort4*)(xq + i) = q;
    }
}

// ---------------- Stage 2: bf16 WMMA GEMM: out[M,N] = xq[M,K] * Wq_t[N,K]^T + bias ----
// Block tile 128x256, 8 waves (2 in M x 4 in N), wave tile 64x64 = 16 WMMA tiles.
// Double-buffered LDS filled by the Tensor Data Mover (TENSORcnt pipeline):
// wave 0 DMAs the A tile, wave 1 the B tile (scalar-gated), all waves compute.
__global__ __launch_bounds__(256) void gemm_bf16_wmma_kernel(
    const unsigned short* __restrict__ Aq,   // M x K bf16
    const unsigned short* __restrict__ Bq,   // N x K bf16 (W^T)
    const float* __restrict__ bias,
    float* __restrict__ out, int M, int N, int K)
{
    __shared__ __align__(16) unsigned short sA[2][BM * KT_PAD];
    __shared__ __align__(16) unsigned short sB[2][BN * KT_PAD];

    const int tid  = threadIdx.x;
    const int lane = tid & 31;
    // wave id as an SGPR: TDM issue gating becomes scalar s_cmp/s_cbranch (no EXEC churn)
    const int wave = __builtin_amdgcn_readfirstlane(tid >> 5);   // 0..7, wave-uniform
    const int wm   = (wave & 1) * 64;     // 2 waves in M (SGPR)
    const int wn   = (wave >> 1) * 64;    // 4 waves in N (SGPR)

    const int bm = blockIdx.y * BM;
    const int bn = blockIdx.x * BN;

    // fragment read mapping (ISA 16-bit A/B layout)
    const int frow = lane & 15;           // row (A) / col (B) within 16x16 tile
    const int kb   = (lane >> 4) * 8;     // lanes 16-31 start at K=8 (and K=24)

    const unsigned ldsA[2] = { lds_off_u32(&sA[0][0]), lds_off_u32(&sA[1][0]) };
    const unsigned ldsB[2] = { lds_off_u32(&sB[0][0]), lds_off_u32(&sB[1][0]) };
    const unsigned long long gA0 = (unsigned long long)(__SIZE_TYPE__)(Aq + (size_t)bm * K);
    const unsigned long long gB0 = (unsigned long long)(__SIZE_TYPE__)(Bq + (size_t)bn * K);

    auto issue_tiles = [&](int k0, int buf) {
        if (wave == 0) {
            tdm_load_2d(ldsA[buf], gA0 + (unsigned long long)k0 * 2,
                        (unsigned)K, (unsigned)M, KT, BM, (unsigned long long)K);
        } else if (wave == 1) {
            tdm_load_2d(ldsB[buf], gB0 + (unsigned long long)k0 * 2,
                        (unsigned)K, (unsigned)N, KT, BN, (unsigned long long)K);
        }
    };

    v8f acc[4][4] = {};

    // prologue: fill buffer 0
    issue_tiles(0, 0);
    __builtin_amdgcn_s_wait_tensorcnt(0);
    __syncthreads();

    int cur = 0;
    for (int k0 = 0; k0 < K; k0 += KT) {
        // start DMA for next K-step into the spare buffer (overlaps with WMMA below)
        if (k0 + KT < K) issue_tiles(k0 + KT, cur ^ 1);

        v16bf afrag[4];
        #pragma unroll
        for (int i = 0; i < 4; ++i) {
            const unsigned short* p = &sA[cur][(wm + i * 16 + frow) * KT_PAD + kb];
            v8bf lo = *(const v8bf*)(p);
            v8bf hi = *(const v8bf*)(p + 16);
            afrag[i] = __builtin_shufflevector(lo, hi, 0,1,2,3,4,5,6,7,8,9,10,11,12,13,14,15);
        }
        v16bf bfrag[4];
        #pragma unroll
        for (int j = 0; j < 4; ++j) {
            const unsigned short* p = &sB[cur][(wn + j * 16 + frow) * KT_PAD + kb];
            v8bf lo = *(const v8bf*)(p);
            v8bf hi = *(const v8bf*)(p + 16);
            bfrag[j] = __builtin_shufflevector(lo, hi, 0,1,2,3,4,5,6,7,8,9,10,11,12,13,14,15);
        }

        #pragma unroll
        for (int i = 0; i < 4; ++i)
            #pragma unroll
            for (int j = 0; j < 4; ++j)
                acc[i][j] = __builtin_amdgcn_wmma_f32_16x16x32_bf16(
                    false, afrag[i], false, bfrag[j],
                    (short)0, acc[i][j], false, false);

        __builtin_amdgcn_s_wait_tensorcnt(0);  // DMA for next buffer has landed
        __syncthreads();                       // visible WG-wide; cur fully consumed
        cur ^= 1;
    }

    // Epilogue: D layout -> VGPR r holds row (r + 8*(lane>>4)), col = lane&15
    const int lrow = 8 * (lane >> 4);
    const int lcol = lane & 15;
    #pragma unroll
    for (int i = 0; i < 4; ++i) {
        #pragma unroll
        for (int j = 0; j < 4; ++j) {
            const int col = bn + wn + j * 16 + lcol;
            const float bb = bias[col];
            #pragma unroll
            for (int r = 0; r < 8; ++r) {
                __builtin_nontemporal_store(acc[i][j][r] + bb,
                    out + (size_t)(bm + wm + i * 16 + lrow + r) * N + col);
            }
        }
    }
}

extern "C" void kernel_launch(void* const* d_in, const int* in_sizes, int n_in,
                              void* d_out, int out_size, void* d_ws, size_t ws_size,
                              hipStream_t stream) {
    const float* x = (const float*)d_in[0];
    const float* W = (const float*)d_in[1];
    const float* b = (const float*)d_in[2];
    float* out = (float*)d_out;

    const int N = in_sizes[2];            // 4096
    const int K = in_sizes[1] / N;        // 4096
    const int M = in_sizes[0] / K;        // 8192

    unsigned short* Wq = (unsigned short*)d_ws;           // N*K bf16
    unsigned short* xq = Wq + (size_t)N * K;              // M*K bf16

    dim3 tb(32, 8);
    dim3 tg(N / 32, K / 32);
    quantize_transpose_kernel<<<tg, tb, 0, stream>>>(W, Wq, K, N);

    long nx = (long)M * K;
    convert_x_kernel<<<(unsigned)((nx / 4 + 255) / 256), 256, 0, stream>>>(x, xq, nx);

    dim3 gg(N / BN, M / BM);
    gemm_bf16_wmma_kernel<<<gg, 256, 0, stream>>>(xq, Wq, b, out, M, N, K);
}
